// Final_Model_57191784513711
// MI455X (gfx1250) — matrix operand order repaired and checked by reading, not compiled, not run on hardware
//
#include <hip/hip_runtime.h>
#include <hip/hip_bf16.h>
#include <math.h>

typedef __attribute__((ext_vector_type(16))) _Float16 v16h;
typedef __attribute__((ext_vector_type(8)))  _Float16 v8h;
typedef __attribute__((ext_vector_type(8)))  float    v8f;
typedef __attribute__((ext_vector_type(4)))  int      v4i;

static constexpr int kB = 1024;
static constexpr int kT = 20;
static constexpr int kC = 512;
static constexpr int kH = 8;
static constexpr int kL = 8;
static constexpr int kD = 64;
static constexpr int kM = kB * kT;          // 20480 token rows
static constexpr size_t kPerLayerW16 =
    (size_t)1536 * 512 + 512 * 512 + 2048 * 512 + 512 * 2048; // 3,145,728 f16

// Async global->LDS staging (CDNA5): guarded so compile never regresses.
#if defined(__gfx1250__) && __has_builtin(__builtin_amdgcn_global_load_async_to_lds_b128)
#define USE_ASYNC_LDS 1
#else
#define USE_ASYNC_LDS 0
#endif

#if USE_ASYNC_LDS
__device__ __forceinline__ void wait_asynccnt0()
{
#if __has_builtin(__builtin_amdgcn_s_wait_asynccnt)
    __builtin_amdgcn_s_wait_asynccnt(0);
#else
    asm volatile("s_wait_asynccnt 0" ::: "memory");
#endif
}
#endif

// ---------------------------------------------------------------------------
// Encoder: h[n,c] = x[n,0]*enc_w[0,c] + x[n,1]*enc_w[1,c] + enc_b[c] + wpe[t+1,c]
// ---------------------------------------------------------------------------
__global__ void enc_kernel(const float* __restrict__ x, const float* __restrict__ ew,
                           const float* __restrict__ eb, const float* __restrict__ wpe,
                           float* __restrict__ Hout)
{
    int idx = blockIdx.x * blockDim.x + threadIdx.x;
    if (idx >= kM * kC) return;
    int n = idx >> 9;
    int c = idx & 511;
    int t = n % kT;
    Hout[idx] = x[2 * n] * ew[c] + x[2 * n + 1] * ew[kC + c] + eb[c]
              + wpe[(size_t)(t + 1) * kC + c];
}

// ---------------------------------------------------------------------------
// Weight convert+transpose: W f32 [K,N] row-major -> Wt f16 [N,K] row-major
// ---------------------------------------------------------------------------
__global__ void cvtT_kernel(const float* __restrict__ W, _Float16* __restrict__ Wt,
                            int K, int N)
{
    int idx = blockIdx.x * blockDim.x + threadIdx.x;
    if (idx >= K * N) return;
    int n = idx / K;
    int k = idx - n * K;
    Wt[idx] = (_Float16)W[(size_t)k * N + n];
}

// ---------------------------------------------------------------------------
// LayerNorm (f32 in, f16 out): one wave32 per row; biased variance
// ---------------------------------------------------------------------------
__global__ __launch_bounds__(128)
void ln_kernel(const float* __restrict__ X, const float* __restrict__ w,
               const float* __restrict__ b, _Float16* __restrict__ Y, int rows)
{
    int row  = blockIdx.x * (blockDim.x >> 5) + (threadIdx.x >> 5);
    int lane = threadIdx.x & 31;
    if (row >= rows) return;
    const float* x = X + (size_t)row * kC;
    float s = 0.f, ss = 0.f;
    for (int c = lane; c < kC; c += 32) {
        float v = x[c];
        s += v; ss += v * v;
    }
    for (int o = 16; o > 0; o >>= 1) {
        s  += __shfl_xor(s,  o, 32);
        ss += __shfl_xor(ss, o, 32);
    }
    const float inv512 = 1.0f / 512.0f;
    float mu  = s * inv512;
    float var = ss * inv512 - mu * mu;
    float inv = rsqrtf(var + 1e-5f);
    _Float16* y = Y + (size_t)row * kC;
    for (int c = lane; c < kC; c += 32)
        y[c] = (_Float16)((x[c] - mu) * inv * w[c] + b[c]);
}

// ---------------------------------------------------------------------------
// helpers
// ---------------------------------------------------------------------------
__device__ __forceinline__ v16h frag_from2(const _Float16* lo8, const _Float16* hi8)
{
    v8h lo = *(const v8h*)lo8;
    v8h hi = *(const v8h*)hi8;
    v16h r;
    #pragma unroll
    for (int i = 0; i < 8; ++i) { r[i] = lo[i]; r[i + 8] = hi[i]; }
    return r;
}

// ---------------------------------------------------------------------------
// WMMA GEMM: Out[M,N] = epilogue(A[M,K](f16) @ Wt^T + bias)
//   Block = 8 waves = 128 M-rows x 64 N-cols. B panel (64 cols x 64 K) staged
//   in LDS (double buffered, async-to-LDS when available) and shared by all
//   8 waves; each wave runs v_wmma_f32_16x16x32_f16 on a 16x64 tile.
//   MODE 0: +bias   MODE 1: +bias+Res   MODE 2: +bias then exact GELU
// ---------------------------------------------------------------------------
template<int MODE, typename OutT>
__global__ __launch_bounds__(256)
void gemm_wmma(const _Float16* __restrict__ A, const _Float16* __restrict__ Bt,
               const float* __restrict__ bias, const float* __restrict__ Res,
               OutT* __restrict__ Out, int M, int N, int K)
{
    __shared__ _Float16 Bs[2][64][72];      // 72 = 64 + pad (bank spread), 16B aligned rows

    const int tid    = threadIdx.x;
    const int waveId = tid >> 5;
    const int lane   = tid & 31;
    const int tilesN = N >> 6;
    const int bm = (blockIdx.x / tilesN) * 128;
    const int wn = (blockIdx.x % tilesN) * 64;
    const int wm = bm + waveId * 16;

    const int rsel = lane >> 4;             // 0: lanes 0-15, 1: lanes 16-31
    const int lidx = lane & 15;
    const _Float16* arow = A + (size_t)(wm + lidx) * K + (rsel << 3);

    const int nch = K >> 6;                 // 64-K chunks

    auto stage = [&](int c, int buf) {
        const int ck = c << 6;
        #pragma unroll
        for (int r = 0; r < 4; ++r) {
            int seg = tid + (r << 8);       // 1024 16B segments per 16KB chunk
            int col = seg >> 3;
            int s8  = (seg & 7) << 3;
            const _Float16* g = Bt + (size_t)(wn + col) * K + ck + s8;
            _Float16* l = &Bs[buf][col][s8];
#if USE_ASYNC_LDS
            __builtin_amdgcn_global_load_async_to_lds_b128(
                (__attribute__((address_space(1))) v4i*)g,
                (__attribute__((address_space(3))) v4i*)l, 0, 0);
#else
            *(v8h*)l = *(const v8h*)g;
#endif
        }
    };

    v8f acc0 = {}, acc1 = {}, acc2 = {}, acc3 = {};

    stage(0, 0);
    for (int c = 0; c < nch; ++c) {
#if USE_ASYNC_LDS
        wait_asynccnt0();
#endif
        __syncthreads();
        if (c + 1 < nch) stage(c + 1, (c + 1) & 1);
        const int buf = c & 1;

        #pragma unroll
        for (int half = 0; half < 2; ++half) {
            const int kk = (c << 6) + (half << 5);
            // A fragment: lanes 0-15 K kk+[0..7],[16..23]; lanes 16-31 +8 (ISA 7.12.2)
            const _Float16* ap = arow + kk;
            v16h a = frag_from2(ap, ap + 16);
            // B fragments from LDS: column wn+j*16+lidx, K [0..7],[16..23] (+8 for lanes 16-31), same as A
            const int kb = (half << 5) + (rsel << 3);
            v16h b0 = frag_from2(&Bs[buf][lidx     ][kb], &Bs[buf][lidx     ][kb + 16]);
            v16h b1 = frag_from2(&Bs[buf][lidx + 16][kb], &Bs[buf][lidx + 16][kb + 16]);
            v16h b2 = frag_from2(&Bs[buf][lidx + 32][kb], &Bs[buf][lidx + 32][kb + 16]);
            v16h b3 = frag_from2(&Bs[buf][lidx + 48][kb], &Bs[buf][lidx + 48][kb + 16]);

            acc0 = __builtin_amdgcn_wmma_f32_16x16x32_f16(false, a, false, b0, (short)0, acc0, false, false);
            acc1 = __builtin_amdgcn_wmma_f32_16x16x32_f16(false, a, false, b1, (short)0, acc1, false, false);
            acc2 = __builtin_amdgcn_wmma_f32_16x16x32_f16(false, a, false, b2, (short)0, acc2, false, false);
            acc3 = __builtin_amdgcn_wmma_f32_16x16x32_f16(false, a, false, b3, (short)0, acc3, false, false);
        }
    }

    v8f accs[4] = {acc0, acc1, acc2, acc3};
    #pragma unroll
    for (int j = 0; j < 4; ++j) {
        int col = wn + j * 16 + lidx;
        float bb = bias[col];
        #pragma unroll
        for (int r = 0; r < 8; ++r) {
            int m = wm + r + (rsel << 3);          // C/D layout: VGPR r -> M=r (+8)
            float v = accs[j][r] + bb;
            if (MODE == 1) v += Res[(size_t)m * N + col];
            if (MODE == 2) v = 0.5f * v * (1.0f + erff(v * 0.70710678118654752f));
            Out[(size_t)m * N + col] = (OutT)v;
        }
    }
}

// ---------------------------------------------------------------------------
// Attention: one wave32 per (b, head). T=20, D=64, causal softmax, fp32 math.
// Reads qkv f32 [M, 3C]; writes f16 y [M, C].
// ---------------------------------------------------------------------------
__global__ __launch_bounds__(32)
void attn_kernel(const float* __restrict__ qkv, _Float16* __restrict__ Y)
{
    __shared__ float qs[kT][kD + 1];
    __shared__ float ks[kT][kD + 1];
    __shared__ float vs[kT][kD + 1];
    __shared__ float ps[32];

    int b  = blockIdx.x / kH;
    int hh = blockIdx.x % kH;
    int lane = threadIdx.x;

    const float* base = qkv + (size_t)b * kT * (3 * kC) + hh * kD;
    for (int i = lane; i < kT * kD; i += 32) {
        int t = i >> 6, d = i & 63;
        const float* p = base + (size_t)t * (3 * kC) + d;
        qs[t][d] = p[0];
        ks[t][d] = p[kC];
        vs[t][d] = p[2 * kC];
    }
    __syncthreads();

    const float scale = 0.125f;          // 1/sqrt(64)
    for (int t = 0; t < kT; ++t) {
        float s = -1e30f;
        if (lane <= t) {
            float acc = 0.f;
            #pragma unroll 8
            for (int d = 0; d < kD; ++d) acc += qs[t][d] * ks[lane][d];
            s = acc * scale;
        }
        float mx = s;
        for (int o = 16; o > 0; o >>= 1) mx = fmaxf(mx, __shfl_xor(mx, o, 32));
        float p = (lane <= t) ? expf(s - mx) : 0.f;
        float sum = p;
        for (int o = 16; o > 0; o >>= 1) sum += __shfl_xor(sum, o, 32);
        ps[lane] = p / sum;
        __syncthreads();

        _Float16* yo = Y + ((size_t)b * kT + t) * kC + hh * kD;
        for (int d = lane; d < kD; d += 32) {
            float acc = 0.f;
            for (int k = 0; k <= t; ++k) acc += ps[k] * vs[k][d];
            yo[d] = (_Float16)acc;
        }
        __syncthreads();
    }
}

// ---------------------------------------------------------------------------
// Predictor: out[n, 0:2] = feat[n,:](f16) @ pred_w[512,2] + pred_b; wave/row
// ---------------------------------------------------------------------------
__global__ __launch_bounds__(128)
void pred_kernel(const _Float16* __restrict__ F, const float* __restrict__ pw,
                 const float* __restrict__ pb, float* __restrict__ Out, int rows)
{
    int row  = blockIdx.x * 4 + (threadIdx.x >> 5);
    int lane = threadIdx.x & 31;
    if (row >= rows) return;
    const _Float16* f = F + (size_t)row * kC;
    float a0 = 0.f, a1 = 0.f;
    for (int c = lane; c < kC; c += 32) {
        float v = (float)f[c];
        a0 += v * pw[2 * c];
        a1 += v * pw[2 * c + 1];
    }
    for (int o = 16; o > 0; o >>= 1) {
        a0 += __shfl_xor(a0, o, 32);
        a1 += __shfl_xor(a1, o, 32);
    }
    if (lane == 0) {
        Out[2 * row]     = a0 + pb[0];
        Out[2 * row + 1] = a1 + pb[1];
    }
}

// ---------------------------------------------------------------------------
template<int MODE, typename OutT>
static void launch_gemm(const _Float16* A, const _Float16* Bt, const float* bias,
                        const float* Res, OutT* Out, int M, int N, int K,
                        hipStream_t s)
{
    int blocks = (M / 128) * (N / 64);
    gemm_wmma<MODE, OutT><<<blocks, 256, 0, s>>>(A, Bt, bias, Res, Out, M, N, K);
}

extern "C" void kernel_launch(void* const* d_in, const int* in_sizes, int n_in,
                              void* d_out, int out_size, void* d_ws, size_t ws_size,
                              hipStream_t stream)
{
    (void)in_sizes; (void)n_in; (void)out_size; (void)ws_size;

    const float* x     = (const float*)d_in[0];
    const float* enc_w = (const float*)d_in[1];
    const float* enc_b = (const float*)d_in[2];
    const float* wpe   = (const float*)d_in[3];
    const float* ln1w  = (const float*)d_in[4];
    const float* ln1b  = (const float*)d_in[5];
    const float* attw  = (const float*)d_in[6];
    const float* attb  = (const float*)d_in[7];
    const float* projw = (const float*)d_in[8];
    const float* projb = (const float*)d_in[9];
    const float* ln2w  = (const float*)d_in[10];
    const float* ln2b  = (const float*)d_in[11];
    const float* f1w   = (const float*)d_in[12];
    const float* f1b   = (const float*)d_in[13];
    const float* f2w   = (const float*)d_in[14];
    const float* f2b   = (const float*)d_in[15];
    const float* lnfw  = (const float*)d_in[16];
    const float* lnfb  = (const float*)d_in[17];
    const float* predw = (const float*)d_in[18];
    const float* predb = (const float*)d_in[19];
    float* out = (float*)d_out;

    // Workspace partition (all 256B aligned)
    char* ws = (char*)d_ws;
    size_t off = 0;
    auto carve = [&](size_t bytes) -> void* {
        void* p = ws + off;
        off += (bytes + 255) & ~(size_t)255;
        return p;
    };
    float*    hbuf  = (float*)carve((size_t)kM * kC * 4);       // residual stream (f32)
    _Float16* tmp16 = (_Float16*)carve((size_t)kM * kC * 2);    // LN out / attn out (f16)
    void*     bigv  = carve((size_t)kM * 3 * kC * 4);           // qkv f32 OR MLP hidden f16
    _Float16* w16   = (_Float16*)carve((size_t)kL * kPerLayerW16 * 2);
    float*    qkvbuf = (float*)bigv;
    _Float16* mid16  = (_Float16*)bigv;                         // [kM, 2048] f16 (aliased)

    // --- Convert & transpose all weights to f16 [N,K] ---
    for (int l = 0; l < kL; ++l) {
        _Float16* lb   = w16 + (size_t)l * kPerLayerW16;
        _Float16* qkvW = lb;
        _Float16* prjW = qkvW + (size_t)1536 * 512;
        _Float16* fc1W = prjW + (size_t)512 * 512;
        _Float16* fc2W = fc1W + (size_t)2048 * 512;
        cvtT_kernel<<<(1536 * 512 + 255) / 256, 256, 0, stream>>>(
            attw  + (size_t)l * 512 * 1536, qkvW, 512, 1536);
        cvtT_kernel<<<(512 * 512 + 255) / 256, 256, 0, stream>>>(
            projw + (size_t)l * 512 * 512,  prjW, 512, 512);
        cvtT_kernel<<<(2048 * 512 + 255) / 256, 256, 0, stream>>>(
            f1w   + (size_t)l * 512 * 2048, fc1W, 512, 2048);
        cvtT_kernel<<<(512 * 2048 + 255) / 256, 256, 0, stream>>>(
            f2w   + (size_t)l * 2048 * 512, fc2W, 2048, 512);
    }

    // --- Encoder + positional embedding ---
    enc_kernel<<<(kM * kC + 255) / 256, 256, 0, stream>>>(x, enc_w, enc_b, wpe, hbuf);

    // --- Transformer blocks ---
    for (int l = 0; l < kL; ++l) {
        _Float16* lb   = w16 + (size_t)l * kPerLayerW16;
        _Float16* qkvW = lb;
        _Float16* prjW = qkvW + (size_t)1536 * 512;
        _Float16* fc1W = prjW + (size_t)512 * 512;
        _Float16* fc2W = fc1W + (size_t)2048 * 512;

        ln_kernel<<<kM / 4, 128, 0, stream>>>(hbuf, ln1w + l * kC, ln1b + l * kC, tmp16, kM);
        launch_gemm<0, float>(tmp16, qkvW, attb + (size_t)l * 3 * kC, nullptr,
                              qkvbuf, kM, 3 * kC, kC, stream);
        attn_kernel<<<kB * kH, 32, 0, stream>>>(qkvbuf, tmp16);
        launch_gemm<1, float>(tmp16, prjW, projb + (size_t)l * kC, hbuf,
                              hbuf, kM, kC, kC, stream);
        ln_kernel<<<kM / 4, 128, 0, stream>>>(hbuf, ln2w + l * kC, ln2b + l * kC, tmp16, kM);
        launch_gemm<2, _Float16>(tmp16, fc1W, f1b + (size_t)l * 4 * kC, nullptr,
                                 mid16, kM, 4 * kC, kC, stream);
        launch_gemm<1, float>(mid16, fc2W, f2b + (size_t)l * kC, hbuf,
                              hbuf, kM, kC, 4 * kC, stream);
    }

    // --- Final LN + predictor ---
    ln_kernel<<<kM / 4, 128, 0, stream>>>(hbuf, lnfw, lnfb, tmp16, kM);
    pred_kernel<<<(kM + 3) / 4, 128, 0, stream>>>(tmp16, predw, predb, out, kM);
}
